// PNN_50723563766466
// MI455X (gfx1250) — matrix operand order repaired and checked by reading, not compiled, not run on hardware
//
#include <hip/hip_runtime.h>
#include <stdint.h>

// ---------- types ----------
typedef _Float16 v16h __attribute__((ext_vector_type(16)));
typedef _Float16 v8h  __attribute__((ext_vector_type(8)));
typedef float    v8f  __attribute__((ext_vector_type(8)));
typedef unsigned int v4u __attribute__((ext_vector_type(4)));
typedef int      v4i  __attribute__((ext_vector_type(4)));
typedef int      v8i  __attribute__((ext_vector_type(8)));

#define BATCH   16384
#define FFIELD  50
#define EDIM    64
#define DDIM    512
#define KFLAT   3200      // F*E
#define FPAD    64
#define H1      1024
#define H2      512
#define BN_EPS  1e-10f

// ---------- WMMA helpers (gfx1250 wave32, 16x16x32 f16 -> f32) ----------
__device__ inline v8f wmma16(v16h a, v16h b, v8f c) {
  return __builtin_amdgcn_wmma_f32_16x16x32_f16(false, a, false, b, (short)0, c,
                                                false, false);
}

// A-matrix 16x32 f16 tile, row-major source, base already at tile row 0.
// lane l (lo=l&15 row, hi=l>>4): halfs {k0+hi*8 .. +8} and {k0+16+hi*8 .. +8}
__device__ inline v16h load_A(const _Float16* base, int lda, int k0, int lane) {
  const int lo = lane & 15, hi = lane >> 4;
  const _Float16* p = base + (size_t)lo * lda + k0 + hi * 8;
  v8h a0 = *(const v8h*)(p);
  v8h a1 = *(const v8h*)(p + 16);
  v16h a;
#pragma unroll
  for (int i = 0; i < 8; ++i) { a[i] = a0[i]; a[i + 8] = a1[i]; }
  return a;
}

// B-matrix 32x16 f16 tile from Bt stored (N,K) row-major, base at tile col row n0.
__device__ inline v16h load_B(const _Float16* baseN, int ldb, int k0, int lane) {
  const int lo = lane & 15, hi = lane >> 4;
  const _Float16* p = baseN + (size_t)lo * ldb + k0 + hi * 8;
  v8h b0 = *(const v8h*)(p);
  v8h b1 = *(const v8h*)(p + 16);
  v16h b;
#pragma unroll
  for (int i = 0; i < 8; ++i) { b[i] = b0[i]; b[i + 8] = b1[i]; }
  return b;
}

// generic -> LDS byte-offset (addrspace(3) ptrtoint)
typedef __attribute__((address_space(3))) char lds_char;
__device__ inline uint32_t lds_offset_of(const void* p) {
  return (uint32_t)(uintptr_t)(lds_char*)p;
}

// Issue a TDM 2D tile load: 64 rows x 128 halfs from (rows,K) f16 matrix -> LDS.
// D# group0/group1 packed per CDNA5 ISA 8.3/8.4 (data_size=1 -> 2 bytes).
__device__ inline void tdm_load_slab(const _Float16* gsrc, uint32_t lds_off, int K) {
  uint64_t ga = (uint64_t)(uintptr_t)gsrc;
  v4u g0;
  g0[0] = 1u;                                   // count=1, user mode
  g0[1] = lds_off;                              // lds_addr (bytes)
  g0[2] = (uint32_t)ga;                         // global_addr[31:0]
  g0[3] = (uint32_t)((ga >> 32) & 0x1FFFFFFu)   // global_addr[56:32]
          | (2u << 30);                         // type=2 (image)
  v8i g1;
  g1[0] = (int)(1u << 16);                      // data_size=1 (2B)
  g1[1] = (int)(128u << 16);                    // tensor_dim0 = 128
  g1[2] = (int)(64u << 16);                     // tensor_dim1 = 64
  g1[3] = (int)(128u << 16);                    // tile_dim0 = 128
  g1[4] = 64;                                   // tile_dim1 = 64
  g1[5] = K;                                    // tensor_dim0_stride[31:0]
  g1[6] = 0;
  g1[7] = 0;
  v4i z4 = {};
  v8i z8 = {};
  __builtin_amdgcn_tensor_load_to_lds(g0, g1, z4, z4, z8, 0);
}

// ---------- prep kernels ----------
__global__ void k_cast_h(const float* __restrict__ src, _Float16* __restrict__ dst, int n) {
  int i = blockIdx.x * 256 + threadIdx.x;
  if (i < n) dst[i] = (_Float16)src[i];
}

__global__ void k_pad_q(const float* __restrict__ q, _Float16* __restrict__ q16) {
  int i = blockIdx.x * 256 + threadIdx.x;           // D*64
  if (i >= DDIM * FPAD) return;
  int f = i & 63, d = i >> 6;
  q16[i] = (f < FFIELD) ? (_Float16)q[d * FFIELD + f] : (_Float16)0.0f;
}

__global__ void k_transpose_h(const float* __restrict__ src, _Float16* __restrict__ dst,
                              int K, int N) {
  int i = blockIdx.x * 256 + threadIdx.x;           // over N*K
  if (i >= K * N) return;
  int k = i % K, n = i / K;
  dst[i] = (_Float16)src[(size_t)k * N + n];
}

// ---------- gather: emb16 (B, F*E) row-major ----------
__global__ void k_gather_rm(const int* __restrict__ idx, const float* __restrict__ fv,
                            const float* __restrict__ tab, _Float16* __restrict__ emb16) {
  int64_t i = (int64_t)blockIdx.x * 256 + threadIdx.x;   // B*F*E
  int e = (int)(i & 63);
  int64_t bf = i >> 6;                                    // b*50+f
  int id = idx[bf];
  float v = fv[bf];
  emb16[i] = (_Float16)(tab[(size_t)id * EDIM + e] * v);
}

// ---------- gather: embT (B, E, Fpad=64) field-major, zero padded ----------
__global__ void k_gather_tr(const int* __restrict__ idx, const float* __restrict__ fv,
                            const float* __restrict__ tab, _Float16* __restrict__ embT) {
  int64_t i = (int64_t)blockIdx.x * 256 + threadIdx.x;   // B*64*64
  int f = (int)(i & 63);
  int64_t be = i >> 6;
  int e = (int)(be & 63);
  int64_t b = be >> 6;
  float val = 0.0f;
  if (f < FFIELD) {
    int64_t bf = b * FFIELD + f;
    val = tab[(size_t)idx[bf] * EDIM + e] * fv[bf];
  }
  embT[i] = (_Float16)val;
}

// ---------- fused product layer: y0 = relu(lz + lp + pbias), f16 (B,512) ----------
// Block = 8 waves as 2(M) x 4(N); wave tile = 32x32 (2x2 WMMA register blocking).
__global__ void k_product(const _Float16* __restrict__ emb16,  // (B,3200)
                          const _Float16* __restrict__ embT,   // (B,64,64)
                          const _Float16* __restrict__ pl16,   // (D,3200)
                          const _Float16* __restrict__ q16,    // (D,64)
                          const float* __restrict__ pbias,     // (D)
                          _Float16* __restrict__ y0) {         // (B,512)
  const int lane = threadIdx.x & 31;
  const int wave = threadIdx.x >> 5;
  const int lo = lane & 15, hi = lane >> 4;
  const int m_w = (wave >> 2) * 32;                // wave row offset in block
  const int n_w = (wave & 3) * 32;                 // wave col offset in block
  const int b0 = blockIdx.x * 64 + m_w;            // 16-row tile pair base (rows)
  const int d0 = blockIdx.y * 128 + n_w;           // 16-col tile pair base (cols)

  // --- lz: (32x3200) x (3200x32), 2x2 blocked ---
  v8f lz[2][2] = {};
  {
    const _Float16* Ab0 = emb16 + (size_t)b0 * KFLAT;
    const _Float16* Ab1 = emb16 + (size_t)(b0 + 16) * KFLAT;
    const _Float16* Bb0 = pl16 + (size_t)d0 * KFLAT;
    const _Float16* Bb1 = pl16 + (size_t)(d0 + 16) * KFLAT;
    for (int k0 = 0; k0 < KFLAT; k0 += 32) {
      v16h a0 = load_A(Ab0, KFLAT, k0, lane);
      v16h a1 = load_A(Ab1, KFLAT, k0, lane);
      v16h w0 = load_B(Bb0, KFLAT, k0, lane);
      v16h w1 = load_B(Bb1, KFLAT, k0, lane);
      lz[0][0] = wmma16(a0, w0, lz[0][0]);
      lz[0][1] = wmma16(a0, w1, lz[0][1]);
      lz[1][0] = wmma16(a1, w0, lz[1][0]);
      lz[1][1] = wmma16(a1, w1, lz[1][1]);
    }
  }

  // --- lp^2: 64 x [(32x64) x (64x32)], q tiles hoisted (e-invariant) ---
  v8f lp2[2][2] = {};
  {
    v16h qb[2][2];  // [nt][kchunk]
#pragma unroll
    for (int nt = 0; nt < 2; ++nt) {
      const _Float16* Qb = q16 + (size_t)(d0 + nt * 16) * FPAD;
      qb[nt][0] = load_B(Qb, FPAD, 0, lane);
      qb[nt][1] = load_B(Qb, FPAD, 32, lane);
    }
    for (int e = 0; e < EDIM; ++e) {
      v16h at[2][2];  // [mt][kchunk]
#pragma unroll
      for (int mt = 0; mt < 2; ++mt) {
        const _Float16* Te = embT + (size_t)(b0 + mt * 16) * (EDIM * FPAD) + e * FPAD;
        at[mt][0] = load_A(Te, EDIM * FPAD, 0, lane);
        at[mt][1] = load_A(Te, EDIM * FPAD, 32, lane);
      }
#pragma unroll
      for (int mt = 0; mt < 2; ++mt)
#pragma unroll
        for (int nt = 0; nt < 2; ++nt) {
          v8f s = {};
          s = wmma16(at[mt][0], qb[nt][0], s);
          s = wmma16(at[mt][1], qb[nt][1], s);
#pragma unroll
          for (int j = 0; j < 8; ++j) lp2[mt][nt][j] += s[j] * s[j];
        }
    }
  }

  // --- epilogue: relu(lz + sqrt(lp2) + bias) -> f16 ---
#pragma unroll
  for (int mt = 0; mt < 2; ++mt)
#pragma unroll
    for (int nt = 0; nt < 2; ++nt) {
      const int d = d0 + nt * 16 + lo;
      const float bs = pbias[d];
#pragma unroll
      for (int j = 0; j < 8; ++j) {
        int m = b0 + mt * 16 + j + 8 * hi;
        float v = lz[mt][nt][j] + sqrtf(lp2[mt][nt][j]) + bs;
        v = v > 0.0f ? v : 0.0f;
        y0[(size_t)m * DDIM + d] = (_Float16)v;
      }
    }
}

// ---------- TDM-staged WMMA GEMM: C f32 (B,N) = A f16 (B,K) * Bt f16 (N,K)^T + bias ----
// Block = 8 waves as 2(M) x 4(N); block tile 64x128; wave tile 32x32.
// A slab (64 rows x 128 halfs) staged into LDS by the Tensor Data Mover,
// double buffered; B (weights) read direct from global (L2 resident).
__global__ void k_gemm_tdm(const _Float16* __restrict__ A, const _Float16* __restrict__ Bt,
                           const float* __restrict__ bias, float* __restrict__ C,
                           int K, int N) {
  __shared__ __align__(128) _Float16 As[2][64 * 128];   // 2 x 16 KB

  const int lane = threadIdx.x & 31;
  const int wave = threadIdx.x >> 5;
  const int lo = lane & 15, hi = lane >> 4;
  const int m_w = (wave >> 2) * 32;
  const int n_w = (wave & 3) * 32;
  const int b0 = blockIdx.x * 64;
  const int n0 = blockIdx.y * 128 + n_w;
  const bool tdm_wave = (wave == 0);               // wave-uniform guard (TDM ignores EXEC)
  const int nchunks = K >> 7;                      // K / 128

  // prologue: stage chunk 0
  if (tdm_wave) {
    tdm_load_slab(A + (size_t)b0 * K, lds_offset_of(&As[0][0]), K);
    __builtin_amdgcn_s_wait_tensorcnt(0);
  }
  __syncthreads();

  v8f acc[2][2] = {};
  int buf = 0;
  for (int c = 0; c < nchunks; ++c) {
    // overlap: issue chunk c+1 into the other buffer while computing chunk c
    if (c + 1 < nchunks && tdm_wave) {
      tdm_load_slab(A + (size_t)b0 * K + (c + 1) * 128,
                    lds_offset_of(&As[buf ^ 1][0]), K);
    }
    const _Float16* As0 = &As[buf][(size_t)(m_w) * 128];
    const _Float16* As1 = &As[buf][(size_t)(m_w + 16) * 128];
    const _Float16* Bb0 = Bt + (size_t)n0 * K + c * 128;
    const _Float16* Bb1 = Bt + (size_t)(n0 + 16) * K + c * 128;
#pragma unroll
    for (int kk = 0; kk < 128; kk += 32) {
      v16h a0 = load_A(As0, 128, kk, lane);
      v16h a1 = load_A(As1, 128, kk, lane);
      v16h w0 = load_B(Bb0, K, kk, lane);
      v16h w1 = load_B(Bb1, K, kk, lane);
      acc[0][0] = wmma16(a0, w0, acc[0][0]);
      acc[0][1] = wmma16(a0, w1, acc[0][1]);
      acc[1][0] = wmma16(a1, w0, acc[1][0]);
      acc[1][1] = wmma16(a1, w1, acc[1][1]);
    }
    if (tdm_wave) __builtin_amdgcn_s_wait_tensorcnt(0);
    __syncthreads();
    buf ^= 1;
  }

#pragma unroll
  for (int mt = 0; mt < 2; ++mt)
#pragma unroll
    for (int nt = 0; nt < 2; ++nt) {
      const int n = n0 + nt * 16 + lo;
      const float bs = bias[n];
#pragma unroll
      for (int j = 0; j < 8; ++j) {
        int m = b0 + m_w + mt * 16 + j + 8 * hi;
        C[(size_t)m * N + n] = acc[mt][nt][j] + bs;
      }
    }
}

// ---------- BN stats: per-column mean / rstd over B=16384 rows ----------
__global__ void k_bn_stats(const float* __restrict__ Z, int N,
                           float* __restrict__ mean, float* __restrict__ rstd) {
  __shared__ float ssum[256];
  __shared__ float ssq[256];
  const int c = blockIdx.x * 64 + (threadIdx.x & 63);
  const int rch = threadIdx.x >> 6;
  float s = 0.0f, s2 = 0.0f;
  for (int r = rch; r < BATCH; r += 4) {
    float x = Z[(size_t)r * N + c];
    s += x;
    s2 += x * x;
  }
  ssum[threadIdx.x] = s;
  ssq[threadIdx.x] = s2;
  __syncthreads();
  if (threadIdx.x < 64) {
    float ts = 0.0f, tq = 0.0f;
#pragma unroll
    for (int i = 0; i < 4; ++i) { ts += ssum[threadIdx.x + 64 * i]; tq += ssq[threadIdx.x + 64 * i]; }
    const float inv = 1.0f / (float)BATCH;
    float m = ts * inv;
    float v = tq * inv - m * m;
    mean[c] = m;
    rstd[c] = rsqrtf(v + BN_EPS);
  }
}

// ---------- normalize + relu + f16 cast ----------
__global__ void k_bn_relu_cast(const float* __restrict__ Z, const float* __restrict__ mean,
                               const float* __restrict__ rstd, const float* __restrict__ g,
                               const float* __restrict__ o, _Float16* __restrict__ Y, int N) {
  int64_t i = (int64_t)blockIdx.x * 256 + threadIdx.x;
  int c = (int)(i % N);
  float y = (Z[i] - mean[c]) * rstd[c] * g[0] + o[0];
  y = y > 0.0f ? y : 0.0f;
  Y[i] = (_Float16)y;
}

// ---------- fused final: BN(z2)+relu, dot with output_w, +b, sigmoid ----------
__global__ void k_output(const float* __restrict__ Z2, const float* __restrict__ mean,
                         const float* __restrict__ rstd, const float* __restrict__ g,
                         const float* __restrict__ o, const float* __restrict__ ow,
                         const float* __restrict__ ob, float* __restrict__ out) {
  __shared__ float red[256];
  const int b = blockIdx.x, t = threadIdx.x;
  float acc = 0.0f;
#pragma unroll
  for (int c = t; c < H2; c += 256) {
    float y = (Z2[(size_t)b * H2 + c] - mean[c]) * rstd[c] * g[0] + o[0];
    y = y > 0.0f ? y : 0.0f;
    acc += y * ow[c];
  }
  red[t] = acc;
  __syncthreads();
  for (int s = 128; s > 0; s >>= 1) {
    if (t < s) red[t] += red[t + s];
    __syncthreads();
  }
  if (t == 0) out[b] = 1.0f / (1.0f + expf(-(red[0] + ob[0])));
}

// ---------- host launcher ----------
extern "C" void kernel_launch(void* const* d_in, const int* in_sizes, int n_in,
                              void* d_out, int out_size, void* d_ws, size_t ws_size,
                              hipStream_t stream) {
  const int*   feat_index = (const int*)d_in[0];
  const float* feat_value = (const float*)d_in[1];
  const float* emb_table  = (const float*)d_in[2];
  const float* prod_lin   = (const float*)d_in[3];
  const float* prod_bias  = (const float*)d_in[4];
  const float* prod_quad  = (const float*)d_in[5];
  const float* layer0     = (const float*)d_in[6];
  const float* bias0      = (const float*)d_in[7];
  const float* layer1     = (const float*)d_in[8];
  const float* bias1      = (const float*)d_in[9];
  const float* bn_scale   = (const float*)d_in[10];
  const float* bn_offset  = (const float*)d_in[11];
  const float* output_w   = (const float*)d_in[12];
  const float* output_b   = (const float*)d_in[13];
  float* out = (float*)d_out;

  // workspace carving (256B aligned regions)
  uint8_t* ws = (uint8_t*)d_ws;
  size_t off = 0;
  auto carve = [&](size_t bytes) -> uint8_t* {
    uint8_t* p = ws + off;
    off = (off + bytes + 255) & ~(size_t)255;
    return p;
  };
  _Float16* emb16 = (_Float16*)carve((size_t)BATCH * KFLAT * 2);        // 104.9 MB
  _Float16* embT  = (_Float16*)carve((size_t)BATCH * EDIM * FPAD * 2);  // 134.2 MB
  _Float16* pl16  = (_Float16*)carve((size_t)DDIM * KFLAT * 2);         // 3.3 MB
  _Float16* q16   = (_Float16*)carve((size_t)DDIM * FPAD * 2);
  _Float16* W0t   = (_Float16*)carve((size_t)H1 * DDIM * 2);            // (N=1024, K=512)
  _Float16* W1t   = (_Float16*)carve((size_t)H2 * H1 * 2);              // (N=512,  K=1024)
  _Float16* y0h   = (_Float16*)carve((size_t)BATCH * DDIM * 2);         // 16.8 MB
  float*    z1    = (float*)carve((size_t)BATCH * H1 * 4);              // 67.1 MB
  float*    mean1 = (float*)carve(H1 * 4);
  float*    rstd1 = (float*)carve(H1 * 4);
  float*    mean2 = (float*)carve(H2 * 4);
  float*    rstd2 = (float*)carve(H2 * 4);
  // dead-region reuse: emb16/embT are dead after k_product
  _Float16* y1h = emb16;                 // 33.6 MB <= 104.9 MB
  float*    z2  = (float*)embT;          // 33.6 MB <= 134.2 MB

  // --- prep: weight casts/layout ---
  {
    int n = DDIM * KFLAT;
    k_cast_h<<<(n + 255) / 256, 256, 0, stream>>>(prod_lin, pl16, n);
  }
  k_pad_q<<<(DDIM * FPAD + 255) / 256, 256, 0, stream>>>(prod_quad, q16);
  k_transpose_h<<<(DDIM * H1 + 255) / 256, 256, 0, stream>>>(layer0, W0t, DDIM, H1);
  k_transpose_h<<<(H1 * H2 + 255) / 256, 256, 0, stream>>>(layer1, W1t, H1, H2);

  // --- gather (table is 25.6 MB -> L2 resident) ---
  k_gather_rm<<<(int)(((int64_t)BATCH * FFIELD * EDIM) / 256), 256, 0, stream>>>(
      feat_index, feat_value, emb_table, emb16);
  k_gather_tr<<<(int)(((int64_t)BATCH * EDIM * FPAD) / 256), 256, 0, stream>>>(
      feat_index, feat_value, emb_table, embT);

  // --- fused product layer (lz + lp + bias + relu) ---
  {
    dim3 grid(BATCH / 64, DDIM / 128);
    k_product<<<grid, 256, 0, stream>>>(emb16, embT, pl16, q16, prod_bias, y0h);
  }

  // --- layer 0 GEMM (TDM-staged) + BN + relu ---
  {
    dim3 grid(BATCH / 64, H1 / 128);
    k_gemm_tdm<<<grid, 256, 0, stream>>>(y0h, W0t, bias0, z1, DDIM, H1);
  }
  k_bn_stats<<<H1 / 64, 256, 0, stream>>>(z1, H1, mean1, rstd1);
  k_bn_relu_cast<<<(int)(((int64_t)BATCH * H1) / 256), 256, 0, stream>>>(
      z1, mean1, rstd1, bn_scale, bn_offset, y1h, H1);

  // --- layer 1 GEMM (TDM-staged) + BN stats ---
  {
    dim3 grid(BATCH / 64, H2 / 128);
    k_gemm_tdm<<<grid, 256, 0, stream>>>(y1h, W1t, bias1, z2, H1, H2);
  }
  k_bn_stats<<<H2 / 64, 256, 0, stream>>>(z2, H2, mean2, rstd2);

  // --- fused BN + relu + output dot + sigmoid ---
  k_output<<<BATCH, 256, 0, stream>>>(z2, mean2, rstd2, bn_scale, bn_offset,
                                      output_w, output_b, out);
}
